// TriangleMultiplicationOutgoing_20418274525423
// MI455X (gfx1250) — matrix operand order repaired and checked by reading, not compiled, not run on hardware
//
#include <hip/hip_runtime.h>
#include <hip/hip_bf16.h>

#define TN  512
#define TC  128
#define TNN (TN*TN)

typedef __attribute__((ext_vector_type(16))) __bf16 v16bf;
typedef __attribute__((ext_vector_type(8)))  __bf16 v8bf;
typedef __attribute__((ext_vector_type(8)))  float  v8f;
typedef __attribute__((ext_vector_type(4)))  float  v4f;
typedef __attribute__((ext_vector_type(4)))  unsigned int v4u;
typedef __attribute__((ext_vector_type(8)))  int    v8i;
typedef __attribute__((ext_vector_type(4)))  int    v4i;

static __device__ __forceinline__ float sigmoidf_(float x){ return 1.0f/(1.0f+__expf(-x)); }

static __device__ __forceinline__ v16bf combine16(v8bf lo, v8bf hi){
  v16bf r;
  #pragma unroll
  for (int i=0;i<8;++i){ r[i]=lo[i]; r[i+8]=hi[i]; }
  return r;
}

// A fragment (16x32 bf16) per ISA layout: lanes 0-15 hold M=lane, K = kk+[0..7],kk+[16..23];
// lanes 16-31 hold M=lane-16, K = kk+[8..15],kk+[24..31].
static __device__ __forceinline__ v16bf frag_a(const __bf16* s, int stride, int lane, int kk){
  const int m  = lane & 15;
  const int k0 = kk + ((lane >> 4) << 3);
  v8bf lo = *(const v8bf*)(s + m*stride + k0);
  v8bf hi = *(const v8bf*)(s + m*stride + k0 + 16);
  return combine16(lo, hi);
}

// B fragment (32x16 bf16): column n is contiguous in K at colptr; lanes 0-15 take K=kk+[0..7],kk+[16..23],
// lanes 16-31 take K=kk+[8..15],kk+[24..31].
static __device__ __forceinline__ v16bf frag_b(const __bf16* colptr, int lane, int kk){
  const int k0 = kk + ((lane >> 4) << 3);
  v8bf lo = *(const v8bf*)(colptr + k0);
  v8bf hi = *(const v8bf*)(colptr + k0 + 16);
  return combine16(lo, hi);
}

// TDM descriptor group1 for a 128-row x 32-col bf16 tile out of a 512x512 bf16 matrix,
// with LDS padding of 16B after every 64B row -> 80B (40-element) LDS row pitch.
static __device__ __forceinline__ v8i tdm_group1(){
  v8i g1;
  g1[0] = (1 << 16)      // data_size = 2 bytes
        | (1 << 20)      // pad_enable
        | (3 << 22)      // pad_interval code 3 = 16 DWORDs (64B) between pads
        | (3 << 25);     // pad_amount  code 3 = 4 DWORDs (16B) of padding
  g1[1] = (int)(512u << 16);   // tensor_dim0 = 512 (bits 79:48)
  g1[2] = (int)(512u << 16);   // tensor_dim1 = 512 (bits 111:80)
  g1[3] = (int)(32u  << 16);   // tile_dim0   = 32  (bits 127:112)
  g1[4] = 128;                 // tile_dim1   = 128 (bits 143:128)
  g1[5] = 512;                 // tensor_dim0_stride = 512 (bits 207:160)
  g1[6] = 0;
  g1[7] = 0;
  return g1;
}

// Issue one TDM Global->LDS tile move (descriptor group0 built per call).
// This toolchain's builtin is the 6-arg form: (g0, g1, g2, g3, extra_group, cpol).
static __device__ __forceinline__ void tdm_load(const __bf16* gsrc, __bf16* ldst, v8i g1){
  const unsigned long long ga = (unsigned long long)(size_t)gsrc;
  v4u g0;
  g0[0] = 1u;                                   // count = 1 valid descriptor
  g0[1] = (unsigned)(size_t)ldst;               // LDS byte address (low 32 bits of generic ptr)
  g0[2] = (unsigned)(ga & 0xFFFFFFFFu);         // global_addr[31:0]
  g0[3] = (unsigned)((ga >> 32) & 0x1FFFFFFu)   // global_addr[56:32]
        | (2u << 30);                           // type = 2 ("image")
  v4i z4 = (v4i){0,0,0,0};
  v8i z8 = (v8i){0,0,0,0,0,0,0,0};
  __builtin_amdgcn_tensor_load_to_lds(g0, g1, z4, z4, z8, 0);
}

// ---------------- weight fp32 -> bf16 stash: order [ga, pa, gb, pb, g, po] ----------------
__global__ __launch_bounds__(256)
void cvt_weights(const float* __restrict__ ga, const float* __restrict__ pa,
                 const float* __restrict__ gb, const float* __restrict__ pb,
                 const float* __restrict__ g,  const float* __restrict__ po,
                 __bf16* __restrict__ wcat)
{
  const int i = blockIdx.x*256 + threadIdx.x;
  const int w = i >> 14, off = i & 16383;
  const float* s;
  switch (w){
    case 0: s = ga; break; case 1: s = pa; break; case 2: s = gb; break;
    case 3: s = pb; break; case 4: s = g;  break; default: s = po; break;
  }
  wcat[i] = (__bf16)s[off];
}

// ---------------- LN(z) + 5 projections, fused gating, transposed a/b epilogue ----------------
__global__ __launch_bounds__(256)
void proj_kernel(const float* __restrict__ z, const float* __restrict__ mask,
                 const float* __restrict__ lnw, const float* __restrict__ lnb,
                 const __bf16* __restrict__ wcat,
                 const float* __restrict__ ga_b, const float* __restrict__ pa_b,
                 const float* __restrict__ gb_b, const float* __restrict__ pb_b,
                 const float* __restrict__ g_b,
                 __bf16* __restrict__ a_t, __bf16* __restrict__ b_t,
                 __bf16* __restrict__ gate)
{
  __shared__ __attribute__((aligned(16))) __bf16 s_zl[16][136];
  const int tid = threadIdx.x;
  const int rowbase = blockIdx.x * 16;

  // LayerNorm: 16 lanes per row, 8 elements each, shfl_xor tree within the 16-lane group.
  {
    const int row = tid >> 4, sub = tid & 15;
    const float* zp = z + (size_t)(rowbase + row) * TC + sub*8;
    v4f a0 = *(const v4f*)(zp);
    v4f a1 = *(const v4f*)(zp + 4);
    float s = 0.f, ss = 0.f;
    #pragma unroll
    for (int e=0;e<4;++e){ s += a0[e]+a1[e]; ss += a0[e]*a0[e] + a1[e]*a1[e]; }
    #pragma unroll
    for (int o=8;o>=1;o>>=1){ s += __shfl_xor(s,o,32); ss += __shfl_xor(ss,o,32); }
    const float mu  = s * (1.0f/TC);
    const float inv = rsqrtf(ss*(1.0f/TC) - mu*mu + 1e-5f);
    #pragma unroll
    for (int e=0;e<8;++e){
      const int k = sub*8 + e;
      const float v = (e<4) ? a0[e] : a1[e-4];
      s_zl[row][k] = (__bf16)(((v - mu)*inv)*lnw[k] + lnb[k]);
    }
  }
  __syncthreads();

  const int wave = tid >> 5, lane = tid & 31;
  const int n = lane & 15, half = lane >> 4;
  const int c = wave*16 + n;                 // this wave owns channels [wave*16, wave*16+16)
  const int m0 = rowbase + half*8;

  v8f acc[5];
  #pragma unroll
  for (int wi=0; wi<5; ++wi){
    v8f a = {};
    const __bf16* wrow = wcat + wi*(TC*TC) + c*TC;
    #pragma unroll
    for (int kk=0; kk<TC; kk+=32){
      v16bf af = frag_a(&s_zl[0][0], 136, lane, kk);
      v16bf bf = frag_b(wrow, lane, kk);
      a = __builtin_amdgcn_wmma_f32_16x16x32_bf16(false, af, false, bf, (short)0, a, false, false);
    }
    acc[wi] = a;
  }

  const float gab = ga_b[c], pab = pa_b[c], gbb = gb_b[c], pbb = pb_b[c], gbias = g_b[c];
  v4f mk0 = *(const v4f*)(mask + m0);
  v4f mk1 = *(const v4f*)(mask + m0 + 4);
  v8bf av, bv;
  #pragma unroll
  for (int r=0;r<8;++r){
    const float mkr = (r<4) ? mk0[r] : mk1[r-4];
    av[r] = (__bf16)(sigmoidf_(acc[0][r]+gab) * (acc[1][r]+pab) * mkr);
    bv[r] = (__bf16)(sigmoidf_(acc[2][r]+gbb) * (acc[3][r]+pbb) * mkr);
  }
  // channel-major (c, row) layout: 8 consecutive rows -> one 16B store per lane
  *(v8bf*)(a_t + (size_t)c*TNN + m0) = av;
  *(v8bf*)(b_t + (size_t)c*TNN + m0) = bv;
  #pragma unroll
  for (int r=0;r<8;++r)
    gate[(size_t)(m0+r)*TC + c] = (__bf16)sigmoidf_(acc[4][r] + gbias);
}

// ---------------- per-channel 512x512x512 GEMM: x[i,j,c] = sum_k a[i,k,c] b[j,k,c] ----------------
// 128x128 block tile, 8 waves in 4x2 grid, each wave: 32x64 region = 2x4 subtiles = 8 WMMAs/step.
// K panels staged Global->LDS by the Tensor Data Mover (double-buffered, wave 0 issues,
// TENSORcnt + workgroup barrier publishes each buffer).
__global__ __launch_bounds__(256)
void tri_kernel(const __bf16* __restrict__ a_t, const __bf16* __restrict__ b_t,
                float* __restrict__ x)
{
  __shared__ __attribute__((aligned(16))) __bf16 sA[2][128][40];
  __shared__ __attribute__((aligned(16))) __bf16 sB[2][128][40];
  const int c  = blockIdx.z;
  const int i0 = blockIdx.y * 128, j0 = blockIdx.x * 128;
  const int tid = threadIdx.x;
  const int wave = tid >> 5, lane = tid & 31;
  const int nlane = lane & 15, half = lane >> 4;
  const int wm = wave >> 1;                 // 0..3 : 32-row strip
  const int wn = wave & 1;                  // 0..1 : 64-col strip
  const bool issuer = (wave == 0);

  const __bf16* Abase = a_t + (size_t)c*TNN + (size_t)i0*TN;   // 128-row A panel
  const __bf16* Bbase = b_t + (size_t)c*TNN + (size_t)j0*TN;   // 128-row B panel
  const v8i g1 = tdm_group1();

  // preload K-tile 0 into buffer 0 via TDM
  if (issuer){
    tdm_load(Abase, &sA[0][0][0], g1);
    tdm_load(Bbase, &sB[0][0][0], g1);
    __builtin_amdgcn_s_wait_tensorcnt(0);
  }
  __syncthreads();

  v8f acc[2][4];
  #pragma unroll
  for (int i=0;i<2;++i)
    #pragma unroll
    for (int j=0;j<4;++j) acc[i][j] = (v8f){};

  const int KSTEPS = TN / 32;               // 16
  for (int it = 0; it < KSTEPS; ++it){
    const int cur = it & 1, nxt = cur ^ 1;

    // kick the DMA for the next K tile before computing (overlaps with WMMAs below)
    if (issuer && (it + 1 < KSTEPS)){
      const int kk = (it + 1) * 32;
      tdm_load(Abase + kk, &sA[nxt][0][0], g1);
      tdm_load(Bbase + kk, &sB[nxt][0][0], g1);
    }

    // compute on current buffer: 2 A-frags x 4 B-frags = 8 WMMAs
    v16bf af0 = frag_a(&sA[cur][wm*32     ][0], 40, lane, 0);
    v16bf af1 = frag_a(&sA[cur][wm*32 + 16][0], 40, lane, 0);
    v16bf bf0 = frag_b(&sB[cur][wn*64      + nlane][0], lane, 0);
    v16bf bf1 = frag_b(&sB[cur][wn*64 + 16 + nlane][0], lane, 0);
    v16bf bf2 = frag_b(&sB[cur][wn*64 + 32 + nlane][0], lane, 0);
    v16bf bf3 = frag_b(&sB[cur][wn*64 + 48 + nlane][0], lane, 0);

    acc[0][0] = __builtin_amdgcn_wmma_f32_16x16x32_bf16(false, af0, false, bf0, (short)0, acc[0][0], false, false);
    acc[0][1] = __builtin_amdgcn_wmma_f32_16x16x32_bf16(false, af0, false, bf1, (short)0, acc[0][1], false, false);
    acc[0][2] = __builtin_amdgcn_wmma_f32_16x16x32_bf16(false, af0, false, bf2, (short)0, acc[0][2], false, false);
    acc[0][3] = __builtin_amdgcn_wmma_f32_16x16x32_bf16(false, af0, false, bf3, (short)0, acc[0][3], false, false);
    acc[1][0] = __builtin_amdgcn_wmma_f32_16x16x32_bf16(false, af1, false, bf0, (short)0, acc[1][0], false, false);
    acc[1][1] = __builtin_amdgcn_wmma_f32_16x16x32_bf16(false, af1, false, bf1, (short)0, acc[1][1], false, false);
    acc[1][2] = __builtin_amdgcn_wmma_f32_16x16x32_bf16(false, af1, false, bf2, (short)0, acc[1][2], false, false);
    acc[1][3] = __builtin_amdgcn_wmma_f32_16x16x32_bf16(false, af1, false, bf3, (short)0, acc[1][3], false, false);

    // publish the next buffer: DMA completion (wave 0) then workgroup barrier
    if (issuer){
      __builtin_amdgcn_s_wait_tensorcnt(0);
    }
    __syncthreads();
  }

  // epilogue: write fp32 x in (i,j,c) layout
  #pragma unroll
  for (int i=0;i<2;++i){
    const int gi = i0 + wm*32 + i*16 + half*8;
    #pragma unroll
    for (int j=0;j<4;++j){
      const int gj = j0 + wn*64 + j*16 + nlane;
      #pragma unroll
      for (int r=0;r<8;++r){
        x[((size_t)(gi + r)*TN + gj)*TC + c] = acc[i][j][r];
      }
    }
  }
}

// ---------------- LN(x) -> po projection -> gate ----------------
__global__ __launch_bounds__(256)
void out_kernel(const float* __restrict__ x, const float* __restrict__ lnw,
                const float* __restrict__ lnb, const __bf16* __restrict__ wcat,
                const float* __restrict__ po_b, const __bf16* __restrict__ gate,
                float* __restrict__ out)
{
  __shared__ __attribute__((aligned(16))) __bf16 s_xl[16][136];
  const int tid = threadIdx.x;
  const int rowbase = blockIdx.x * 16;
  {
    const int row = tid >> 4, sub = tid & 15;
    const float* xp = x + (size_t)(rowbase + row) * TC + sub*8;
    v4f a0 = *(const v4f*)(xp);
    v4f a1 = *(const v4f*)(xp + 4);
    float s = 0.f, ss = 0.f;
    #pragma unroll
    for (int e=0;e<4;++e){ s += a0[e]+a1[e]; ss += a0[e]*a0[e] + a1[e]*a1[e]; }
    #pragma unroll
    for (int o=8;o>=1;o>>=1){ s += __shfl_xor(s,o,32); ss += __shfl_xor(ss,o,32); }
    const float mu  = s * (1.0f/TC);
    const float inv = rsqrtf(ss*(1.0f/TC) - mu*mu + 1e-5f);
    #pragma unroll
    for (int e=0;e<8;++e){
      const int k = sub*8 + e;
      const float v = (e<4) ? a0[e] : a1[e-4];
      s_xl[row][k] = (__bf16)(((v - mu)*inv)*lnw[k] + lnb[k]);
    }
  }
  __syncthreads();

  const int wave = tid >> 5, lane = tid & 31;
  const int n = lane & 15, half = lane >> 4;
  const int c = wave*16 + n;
  const int m0 = rowbase + half*8;

  v8f acc = {};
  const __bf16* wrow = wcat + 5*(TC*TC) + c*TC;   // po weights
  #pragma unroll
  for (int kk=0; kk<TC; kk+=32){
    v16bf af = frag_a(&s_xl[0][0], 136, lane, kk);
    v16bf bf = frag_b(wrow, lane, kk);
    acc = __builtin_amdgcn_wmma_f32_16x16x32_bf16(false, af, false, bf, (short)0, acc, false, false);
  }
  const float pb_ = po_b[c];
  #pragma unroll
  for (int r=0;r<8;++r){
    const float g = (float)gate[(size_t)(m0+r)*TC + c];
    out[(size_t)(m0+r)*TC + c] = g * (acc[r] + pb_);
  }
}

extern "C" void kernel_launch(void* const* d_in, const int* in_sizes, int n_in,
                              void* d_out, int out_size, void* d_ws, size_t ws_size,
                              hipStream_t stream)
{
  const float* z     = (const float*)d_in[0];
  const float* mask  = (const float*)d_in[1];
  const float* lni_w = (const float*)d_in[2];
  const float* lni_b = (const float*)d_in[3];
  const float* lno_w = (const float*)d_in[4];
  const float* lno_b = (const float*)d_in[5];
  const float* pa_w  = (const float*)d_in[6];
  const float* pa_b  = (const float*)d_in[7];
  const float* ga_w  = (const float*)d_in[8];
  const float* ga_b  = (const float*)d_in[9];
  const float* pb_w  = (const float*)d_in[10];
  const float* pb_b  = (const float*)d_in[11];
  const float* gb_w  = (const float*)d_in[12];
  const float* gb_b  = (const float*)d_in[13];
  const float* g_w   = (const float*)d_in[14];
  const float* g_b   = (const float*)d_in[15];
  const float* po_w  = (const float*)d_in[16];
  const float* po_b  = (const float*)d_in[17];
  float* out = (float*)d_out;

  // workspace carve-up (256B aligned)
  char* ws = (char*)d_ws;
  auto carve = [&](size_t bytes)->char* {
    char* p = ws;
    ws += (bytes + 255) & ~(size_t)255;
    return p;
  };
  __bf16* wcat  = (__bf16*)carve((size_t)6*TC*TC*sizeof(__bf16));
  __bf16* a_t   = (__bf16*)carve((size_t)TC*TNN*sizeof(__bf16));
  __bf16* b_t   = (__bf16*)carve((size_t)TC*TNN*sizeof(__bf16));
  __bf16* gateb = (__bf16*)carve((size_t)TNN*TC*sizeof(__bf16));
  float*  x     = (float*) carve((size_t)TNN*TC*sizeof(float));

  cvt_weights<<<384, 256, 0, stream>>>(ga_w, pa_w, gb_w, pb_w, g_w, po_w, wcat);
  proj_kernel<<<TNN/16, 256, 0, stream>>>(z, mask, lni_w, lni_b, wcat,
                                          ga_b, pa_b, gb_b, pb_b, g_b,
                                          a_t, b_t, gateb);
  tri_kernel<<<dim3(TN/128, TN/128, TC), 256, 0, stream>>>(a_t, b_t, x);
  out_kernel<<<TNN/16, 256, 0, stream>>>(x, lno_w, lno_b, wcat, po_b, gateb, out);
}
